// MicroKineticEncoder_15556371546932
// MI455X (gfx1250) — matrix operand order repaired and checked by reading, not compiled, not run on hardware
//
#include <hip/hip_runtime.h>
#include <hip/hip_bf16.h>

// ---------------- types ----------------
typedef __bf16 bf16;
typedef __attribute__((ext_vector_type(16))) __bf16 v16bf;
typedef __attribute__((ext_vector_type(8)))  __bf16 bf16x8;
typedef __attribute__((ext_vector_type(8)))  float  v8f;

#define B_     32
#define T_     4096
#define DIN_   768
#define CC_    256
#define DM_    256
#define KMAX_  32
#define NTYPES_ 16
#define NSCAL_  8
#define EPS_   1e-5f

// output flat offsets (reference tuple order, all as float)
#define OFF_TOK 0                               // [32][32][256]
#define OFF_EM  (OFF_TOK + B_*KMAX_*DM_)        // [32][32]
#define OFF_TP  (OFF_EM  + B_*KMAX_)            // [32][32]
#define OFF_TY  (OFF_TP  + B_*KMAX_)            // [32][32]
#define OFF_CF  (OFF_TY  + B_*KMAX_)            // [32][32]
#define OFF_SC  (OFF_CF  + B_*KMAX_)            // [32][32][8]
#define OFF_DE  (OFF_SC  + B_*KMAX_*NSCAL_)     // [32][32]

__device__ __forceinline__ float gelu_exact(float x) {
    return 0.5f * x * (1.0f + erff(x * 0.70710678118654752440f));
}
__device__ __forceinline__ float sigmoid_f(float x) {
    return 1.0f / (1.0f + expf(-x));
}
__device__ __forceinline__ bf16x8 zero8h() {
    bf16x8 z = {(bf16)0.f, (bf16)0.f, (bf16)0.f, (bf16)0.f,
                (bf16)0.f, (bf16)0.f, (bf16)0.f, (bf16)0.f};
    return z;
}

// ---------------- async global->LDS copy (gfx1250), with sync fallback ----------------
#if __has_builtin(__builtin_amdgcn_global_load_async_to_lds_b128)
#define HAS_ASYNC_LDS 1
#else
#define HAS_ASYNC_LDS 0
#endif

// builtin parameter type (from clang diagnostic): 'int __vector(4)' pointers,
// global (AS1) source and LDS (AS3) destination.
typedef int v4i_ __attribute__((vector_size(4 * sizeof(int))));
typedef __attribute__((address_space(1))) v4i_* gp128_t;
typedef __attribute__((address_space(3))) v4i_* lp128_t;

// copy 16 bytes global -> LDS
__device__ __forceinline__ void cp_async16(void* lds, const void* g) {
#if HAS_ASYNC_LDS
    __builtin_amdgcn_global_load_async_to_lds_b128((gp128_t)(void*)g, (lp128_t)lds, 0, 0);
#else
    *(uint4*)lds = *(const uint4*)g;
#endif
}

template <int N>
__device__ __forceinline__ void wait_async() {
#if HAS_ASYNC_LDS
#if __has_builtin(__builtin_amdgcn_s_wait_asynccnt)
    __builtin_amdgcn_s_wait_asynccnt(N);
#else
    asm volatile("s_wait_asynccnt %0" ::"i"(N) : "memory");
#endif
#endif
}

// ---- WMMA fragment loaders (wave32, 16x16x32 bf16) ----
// A 16x32 bf16: lanes 0-15 -> row M=lane, K 0..15;
//               lanes 16-31 -> row M=lane-16, K 16..31.
__device__ __forceinline__ v16bf load_a_frag(const bf16* As, int stride, int mt, int lane) {
    const int row = mt * 16 + (lane & 15);
    const int kh  = lane >> 4;
    const bf16* p = As + row * stride + kh * 16;
    bf16x8 lo = *(const bf16x8*)p;         // K = kh*16 .. +7
    bf16x8 hi = *(const bf16x8*)(p + 8);   // K = kh*16+8 .. +7
    v16bf a;
#pragma unroll
    for (int i = 0; i < 8; ++i) { a[i] = lo[i]; a[i + 8] = hi[i]; }
    return a;
}
// B 32x16 bf16 stored in LDS transposed as Bs[col][K] (K contiguous per col):
// lanes 0-15 -> col N=lane, rows K=0..15; lanes 16-31 -> col N=lane-16, rows K=16..31.
__device__ __forceinline__ v16bf load_b_frag(const bf16* Bs, int nt, int lane) {
    const int col = lane & 15;
    const int kh  = lane >> 4;
    return *(const v16bf*)(Bs + (size_t)(nt * 16 + col) * 32 + kh * 16);
}

// convert 8 f32 -> 8 bf16, single 16B store
__device__ __forceinline__ void cvt_store8(bf16* dst, const float* src) {
    float4 a = *(const float4*)src;
    float4 b = *(const float4*)(src + 4);
    bf16x8 h;
    h[0] = (bf16)a.x; h[1] = (bf16)a.y; h[2] = (bf16)a.z; h[3] = (bf16)a.w;
    h[4] = (bf16)b.x; h[5] = (bf16)b.y; h[6] = (bf16)b.z; h[7] = (bf16)b.w;
    *(bf16x8*)dst = h;
}

// ---------------- pre-transform kernels ----------------
// features f32 [B][T][DIN] -> bf16 same layout
__global__ __launch_bounds__(256) void k_x2bf(const float* __restrict__ in,
                                              bf16* __restrict__ out) {
    size_t i = ((size_t)blockIdx.x * 256 + threadIdx.x) * 8;
    cvt_store8(&out[i], &in[i]);
}

// conv weights f32 [CC][DIN][k] -> bf16 transposed [slab=(br,dt)][co][ci]
__global__ __launch_bounds__(256) void k_conv_wT(
        const float* __restrict__ w0, const float* __restrict__ w1,
        const float* __restrict__ w2, bf16* __restrict__ out) {
    int i = blockIdx.x * 256 + threadIdx.x;
    if (i >= 15 * CC_ * DIN_) return;
    int ci = i % DIN_;
    int rest = i / DIN_;
    int co = rest & 255;
    int s  = rest >> 8;  // slab 0..14
    int dt, k;
    const float* w;
    if (s < 3)      { w = w0; dt = s;     k = 3; }
    else if (s < 8) { w = w1; dt = s - 3; k = 5; }
    else            { w = w2; dt = s - 8; k = 7; }
    out[i] = (bf16)w[((size_t)co * DIN_ + ci) * k + dt];
}

// fuse_w f32 [768][256] -> bf16 [256 out][768 in]
__global__ __launch_bounds__(256) void k_fuse_wT(const float* __restrict__ w,
                                                 bf16* __restrict__ out) {
    int i = blockIdx.x * 256 + threadIdx.x;
    if (i >= CC_ * 3 * CC_) return;
    int ci = i % (3 * CC_);
    int o  = i / (3 * CC_);
    out[i] = (bf16)w[(size_t)ci * CC_ + o];
}

// gate_w1 f32 [256][64] -> bf16 [64 out][256 in]
__global__ __launch_bounds__(256) void k_g1_wT(const float* __restrict__ w,
                                               bf16* __restrict__ out) {
    int i = blockIdx.x * 256 + threadIdx.x;
    if (i >= 64 * CC_) return;
    int ci = i % CC_;
    int o  = i / CC_;
    out[i] = (bf16)w[(size_t)ci * 64 + o];
}

// ---------------- conv (implicit GEMM, WMMA, async double-buffer) ----------------
// grid: (T/64, B, 3). One WG -> 64 t-rows x 256 channels of one branch.
__global__ __launch_bounds__(256) void k_conv_wmma(
        const bf16* __restrict__ Xbf,           // [B][T][DIN] bf16
        const bf16* __restrict__ wconvT,        // [15][CC][DIN] bf16 (transposed)
        const float* __restrict__ cb0, const float* __restrict__ cb1,
        const float* __restrict__ cb2,
        bf16* __restrict__ Ybf,                 // [3][B][T][CC]
        float* __restrict__ gsum, float* __restrict__ gsq) {  // [3][B][16]
    const int br = blockIdx.z;
    const int b  = blockIdx.y;
    const int t0 = blockIdx.x * 64;
    const int ksz   = (br == 0) ? 3 : ((br == 1) ? 5 : 7);
    const int pad   = ksz >> 1;
    const int slab0 = (br == 0) ? 0 : ((br == 1) ? 3 : 8);
    const float* bias = (br == 0) ? cb0 : ((br == 1) ? cb1 : cb2);

    __shared__ __align__(16) bf16 As[2][64 * 40];   // rows x 32K (stride 40 halves)
    __shared__ __align__(32) bf16 Bs[2][256 * 32];  // [col][K]

    const int tid  = threadIdx.x;
    const int lane = tid & 31;
    const int wave = tid >> 5;
    const int mt   = wave & 3;   // M sub-tile 0..3
    const int ng   = wave >> 2;  // 0..1 -> N tiles ng*8..+7

    const v8f zero8 = {0.f, 0.f, 0.f, 0.f, 0.f, 0.f, 0.f, 0.f};
    v8f acc[8];
#pragma unroll
    for (int i = 0; i < 8; ++i) acc[i] = zero8;

    const int mrow = tid >> 2;   // staging row 0..63
    const int part = tid & 3;    // staging 8-element K segment

    const int nch = ksz * (DIN_ / 32);

    auto issue = [&](int c, int bufi) {
        const int dt = c / 24;
        const int kc = (c - dt * 24) << 5;
        // A: shifted bf16 features, 16B per thread (5 async per wave per chunk total)
        const int tin = t0 + mrow + dt - pad;
        bf16* adst = &As[bufi][mrow * 40 + part * 8];
        if (tin >= 0 && tin < T_) {
            cp_async16(adst, Xbf + ((size_t)b * T_ + tin) * DIN_ + kc + part * 8);
        } else {
            *(bf16x8*)adst = zero8h();
        }
        // B: transposed weight rows, 64B per thread (4 x 16B)
        const bf16* bsrc = wconvT + ((size_t)(slab0 + dt) * CC_ + tid) * DIN_ + kc;
        bf16* bdst = &Bs[bufi][tid * 32];
        cp_async16(bdst +  0, bsrc +  0);
        cp_async16(bdst +  8, bsrc +  8);
        cp_async16(bdst + 16, bsrc + 16);
        cp_async16(bdst + 24, bsrc + 24);
    };
    auto compute = [&](int bufi) {
        v16bf a = load_a_frag(As[bufi], 40, mt, lane);
#pragma unroll
        for (int n0 = 0; n0 < 8; ++n0) {
            v16bf bfr = load_b_frag(Bs[bufi], ng * 8 + n0, lane);
            acc[n0] = __builtin_amdgcn_wmma_f32_16x16x32_bf16(
                false, a, false, bfr, (short)0, acc[n0], false, false);
        }
    };

    issue(0, 0);
    for (int c = 0; c < nch; ++c) {
        const int cur = c & 1;
        if (c + 1 < nch) { issue(c + 1, cur ^ 1); wait_async<5>(); }
        else             { wait_async<0>(); }
        __syncthreads();
        compute(cur);
        __syncthreads();
    }

    // epilogue: +bias, store bf16, per-group (== per n-tile) stats
#pragma unroll
    for (int n0 = 0; n0 < 8; ++n0) {
        const int nt  = ng * 8 + n0;             // group id (16 ch per group)
        const int col = nt * 16 + (lane & 15);
        const float bv = bias[col];
        float s = 0.f, q = 0.f;
#pragma unroll
        for (int j = 0; j < 8; ++j) {
            const int row = mt * 16 + j + ((lane >> 4) << 3);
            float v = acc[n0][j] + bv;
            s += v; q += v * v;
            Ybf[(((size_t)br * B_ + b) * T_ + (t0 + row)) * CC_ + col] = (bf16)v;
        }
#pragma unroll
        for (int off = 16; off > 0; off >>= 1) {  // wave32 reduction
            s += __shfl_xor(s, off, 32);
            q += __shfl_xor(q, off, 32);
        }
        if (lane == 0) {
            atomicAdd(&gsum[((size_t)br * B_ + b) * 16 + nt], s);
            atomicAdd(&gsq [((size_t)br * B_ + b) * 16 + nt], q);
        }
    }
}

// ---------------- GroupNorm finalize + exact GELU (in place, 8 elems/thread) ----------------
__global__ __launch_bounds__(256) void k_gn_gelu(
        bf16* __restrict__ Ybf, const float* __restrict__ gsum,
        const float* __restrict__ gsq,
        const float* __restrict__ g0, const float* __restrict__ be0,
        const float* __restrict__ g1, const float* __restrict__ be1,
        const float* __restrict__ g2, const float* __restrict__ be2) {
    size_t i = ((size_t)blockIdx.x * 256 + threadIdx.x) * 8;
    const float inv_cnt = 1.0f / (16.0f * (float)T_);
    const int ch  = (int)(i & 255);          // 8-aligned -> whole block in one group
    const size_t r2 = i >> 20;               // / (256*4096)
    const int b   = (int)(r2 & 31);
    const int brx = (int)(r2 >> 5);
    const int grp = ch >> 4;
    const size_t si = ((size_t)brx * B_ + b) * 16 + grp;
    const float mu  = gsum[si] * inv_cnt;
    const float var = gsq[si] * inv_cnt - mu * mu;
    const float rstd = rsqrtf(var + EPS_);
    const float* gg = (brx == 0) ? g0  : ((brx == 1) ? g1  : g2);
    const float* bb = (brx == 0) ? be0 : ((brx == 1) ? be1 : be2);
    bf16x8 x = *(const bf16x8*)&Ybf[i];
    bf16x8 y;
#pragma unroll
    for (int e = 0; e < 8; ++e) {
        float xv = (float)x[e];
        float yv = (xv - mu) * rstd * gg[ch + e] + bb[ch + e];
        y[e] = (bf16)gelu_exact(yv);
    }
    *(bf16x8*)&Ybf[i] = y;
}

// ---------------- fuse GEMM (768->256) + GELU + row LayerNorm ----------------
// grid: (T/64, B). WG covers full 256-wide rows, so LN fuses in epilogue.
__global__ __launch_bounds__(256) void k_fuse_wmma(
        const bf16* __restrict__ Ybf,     // [3][B][T][CC] (post GN+GELU)
        const bf16* __restrict__ wfT,     // [256 out][768 in] bf16
        const float* __restrict__ fb, const float* __restrict__ lng,
        const float* __restrict__ lnb,
        float* __restrict__ fusedf, bf16* __restrict__ fusedbf) {
    const int b  = blockIdx.y;
    const int t0 = blockIdx.x * 64;

    __shared__ __align__(16) bf16 As[2][64 * 40];
    __shared__ __align__(32) bf16 Bs[2][256 * 32];
    __shared__ float Cf[64 * 260];
    __shared__ float rs[64][4], rq[64][4];
    __shared__ float rmu[64], rrstd[64];

    const int tid  = threadIdx.x;
    const int lane = tid & 31;
    const int wave = tid >> 5;
    const int mt   = wave & 3;
    const int ng   = wave >> 2;

    const v8f zero8 = {0.f, 0.f, 0.f, 0.f, 0.f, 0.f, 0.f, 0.f};
    v8f acc[8];
#pragma unroll
    for (int i = 0; i < 8; ++i) acc[i] = zero8;

    const int mrow = tid >> 2;
    const int part = tid & 3;

    auto issue = [&](int c, int bufi) {
        const int kc = c << 5;
        const int ch768 = kc + part * 8;    // 8-seg never crosses 256-boundary
        const int brc = ch768 >> 8;
        const int ch  = ch768 & 255;
        cp_async16(&As[bufi][mrow * 40 + part * 8],
                   Ybf + (((size_t)brc * B_ + b) * T_ + (t0 + mrow)) * CC_ + ch);
        const bf16* bsrc = wfT + (size_t)tid * (3 * CC_) + kc;
        bf16* bdst = &Bs[bufi][tid * 32];
        cp_async16(bdst +  0, bsrc +  0);
        cp_async16(bdst +  8, bsrc +  8);
        cp_async16(bdst + 16, bsrc + 16);
        cp_async16(bdst + 24, bsrc + 24);
    };
    auto compute = [&](int bufi) {
        v16bf a = load_a_frag(As[bufi], 40, mt, lane);
#pragma unroll
        for (int n0 = 0; n0 < 8; ++n0) {
            v16bf bfr = load_b_frag(Bs[bufi], ng * 8 + n0, lane);
            acc[n0] = __builtin_amdgcn_wmma_f32_16x16x32_bf16(
                false, a, false, bfr, (short)0, acc[n0], false, false);
        }
    };

    const int nch = 3 * CC_ / 32;  // 24
    issue(0, 0);
    for (int c = 0; c < nch; ++c) {
        const int cur = c & 1;
        if (c + 1 < nch) { issue(c + 1, cur ^ 1); wait_async<5>(); }
        else             { wait_async<0>(); }
        __syncthreads();
        compute(cur);
        __syncthreads();
    }

    // epilogue: +bias, GELU -> LDS
#pragma unroll
    for (int n0 = 0; n0 < 8; ++n0) {
        const int nt  = ng * 8 + n0;
        const int col = nt * 16 + (lane & 15);
        const float bv = fb[col];
#pragma unroll
        for (int j = 0; j < 8; ++j) {
            const int row = mt * 16 + j + ((lane >> 4) << 3);
            Cf[row * 260 + col] = gelu_exact(acc[n0][j] + bv);
        }
    }
    __syncthreads();

    // row LayerNorm over 256
    {
        const int r = tid >> 2, sub = tid & 3;
        float s = 0.f, q = 0.f;
        for (int c = sub * 64; c < sub * 64 + 64; ++c) {
            float v = Cf[r * 260 + c];
            s += v; q += v * v;
        }
        rs[r][sub] = s; rq[r][sub] = q;
    }
    __syncthreads();
    if ((tid & 3) == 0) {
        const int r = tid >> 2;
        float s = rs[r][0] + rs[r][1] + rs[r][2] + rs[r][3];
        float q = rq[r][0] + rq[r][1] + rq[r][2] + rq[r][3];
        float mu = s * (1.0f / 256.0f);
        float var = q * (1.0f / 256.0f) - mu * mu;
        rmu[r] = mu; rrstd[r] = rsqrtf(var + EPS_);
    }
    __syncthreads();
    for (int e = 0; e < 64; ++e) {
        const int linear = e * 256 + tid;
        const int r = linear >> 8, c = linear & 255;
        float y = (Cf[r * 260 + c] - rmu[r]) * rrstd[r] * lng[c] + lnb[c];
        size_t o = ((size_t)b * T_ + t0 + r) * CC_ + c;
        fusedf[o]  = y;
        fusedbf[o] = (bf16)y;
    }
}

// ---------------- gate GEMM (256->64) + GELU + (64->1) + mask + sigmoid ----------------
// grid: (T/64, B)
__global__ __launch_bounds__(256) void k_gate_wmma(
        const bf16* __restrict__ fusedbf, const bf16* __restrict__ wg1T,  // [64 out][256 in]
        const float* __restrict__ gb1, const float* __restrict__ w2,      // [64]
        const float* __restrict__ b2, const unsigned char* __restrict__ mask,
        float* __restrict__ scores) {
    const int b  = blockIdx.y;
    const int t0 = blockIdx.x * 64;

    __shared__ __align__(16) bf16 As[2][64 * 40];
    __shared__ __align__(32) bf16 Bs[2][64 * 32];
    __shared__ float Hs[64 * 72];
    __shared__ float ps[64][4];

    const int tid  = threadIdx.x;
    const int lane = tid & 31;
    const int wave = tid >> 5;
    const int mt   = wave & 3;
    const int nb   = wave >> 2;   // N tiles nb*2, nb*2+1

    const v8f zero8 = {0.f, 0.f, 0.f, 0.f, 0.f, 0.f, 0.f, 0.f};
    v8f acc[2] = {zero8, zero8};

    const int mrow = tid >> 2;
    const int part = tid & 3;

    auto issue = [&](int c, int bufi) {
        const int kc = c << 5;
        cp_async16(&As[bufi][mrow * 40 + part * 8],
                   fusedbf + ((size_t)b * T_ + t0 + mrow) * CC_ + kc + part * 8);
        // B: 64 cols x 64B, 16B per thread
        cp_async16(&Bs[bufi][mrow * 32 + part * 8],
                   wg1T + (size_t)mrow * CC_ + kc + part * 8);
    };
    auto compute = [&](int bufi) {
        v16bf a = load_a_frag(As[bufi], 40, mt, lane);
#pragma unroll
        for (int i = 0; i < 2; ++i) {
            v16bf bfr = load_b_frag(Bs[bufi], nb * 2 + i, lane);
            acc[i] = __builtin_amdgcn_wmma_f32_16x16x32_bf16(
                false, a, false, bfr, (short)0, acc[i], false, false);
        }
    };

    const int nch = CC_ / 32;  // 8
    issue(0, 0);
    for (int c = 0; c < nch; ++c) {
        const int cur = c & 1;
        if (c + 1 < nch) { issue(c + 1, cur ^ 1); wait_async<2>(); }
        else             { wait_async<0>(); }
        __syncthreads();
        compute(cur);
        __syncthreads();
    }

#pragma unroll
    for (int i = 0; i < 2; ++i) {
        const int nt  = nb * 2 + i;
        const int col = nt * 16 + (lane & 15);
        const float bv = gb1[col];
#pragma unroll
        for (int j = 0; j < 8; ++j) {
            const int row = mt * 16 + j + ((lane >> 4) << 3);
            Hs[row * 72 + col] = gelu_exact(acc[i][j] + bv);
        }
    }
    __syncthreads();

    const int r = tid >> 2, sub = tid & 3;
    float s = 0.f;
    for (int c = sub * 16; c < sub * 16 + 16; ++c) s += Hs[r * 72 + c] * w2[c];
    ps[r][sub] = s;
    __syncthreads();
    if (sub == 0) {
        float gate = ps[r][0] + ps[r][1] + ps[r][2] + ps[r][3] + b2[0];
        const size_t ti = (size_t)b * T_ + t0 + r;
        scores[ti] = mask[ti] ? sigmoid_f(gate) : 0.0f;  // sigmoid(-inf) = 0
    }
}

// ---------------- top-k (K=32) per batch, indices sorted ascending ----------------
__global__ __launch_bounds__(256) void k_topk(const float* __restrict__ scores,
                                              int* __restrict__ topi,
                                              float* __restrict__ tops) {
    const int b = blockIdx.x;
    __shared__ float sc[T_];
    __shared__ float bv[256];
    __shared__ int   bi[256];
    __shared__ int   sel_i[KMAX_];
    __shared__ float sel_s[KMAX_];
    for (int i = threadIdx.x; i < T_; i += 256) sc[i] = scores[(size_t)b * T_ + i];
    __syncthreads();
    for (int it = 0; it < KMAX_; ++it) {
        float best = -1e30f; int besti = 0x7fffffff;
        for (int i = threadIdx.x; i < T_; i += 256) {
            float v = sc[i];
            if (v > best) { best = v; besti = i; }   // earliest index kept on ties
        }
        bv[threadIdx.x] = best; bi[threadIdx.x] = besti;
        __syncthreads();
        if (threadIdx.x == 0) {
            float bb = -1e30f; int ii = 0x7fffffff;
            for (int j = 0; j < 256; ++j) {
                if (bv[j] > bb || (bv[j] == bb && bi[j] < ii)) { bb = bv[j]; ii = bi[j]; }
            }
            sel_i[it] = ii; sel_s[it] = bb; sc[ii] = -1e30f;
        }
        __syncthreads();
    }
    if (threadIdx.x == 0) {  // sort selected by index ascending
        for (int i = 1; i < KMAX_; ++i) {
            int ki = sel_i[i]; float ks = sel_s[i]; int j = i - 1;
            while (j >= 0 && sel_i[j] > ki) { sel_i[j+1] = sel_i[j]; sel_s[j+1] = sel_s[j]; --j; }
            sel_i[j+1] = ki; sel_s[j+1] = ks;
        }
    }
    __syncthreads();
    if (threadIdx.x < KMAX_) {
        topi[b * KMAX_ + threadIdx.x] = sel_i[threadIdx.x];
        tops[b * KMAX_ + threadIdx.x] = sel_s[threadIdx.x];
    }
}

// ---------------- event heads ----------------
// grid: (KMAX, B), 256 threads
__global__ __launch_bounds__(256) void k_head(
        const float* __restrict__ fusedf, const int* __restrict__ topi,
        const unsigned char* __restrict__ mask,
        const float* __restrict__ tokw, const float* __restrict__ tokb,
        const float* __restrict__ tlng, const float* __restrict__ tlnb,
        const float* __restrict__ scw, const float* __restrict__ scb,
        const float* __restrict__ tyw, const float* __restrict__ tyb,
        const float* __restrict__ cfw, const float* __restrict__ cfb,
        float* __restrict__ out) {
    const int b = blockIdx.y, k = blockIdx.x;
    const int tid = threadIdx.x;
    __shared__ float ev[CC_];
    __shared__ float red[256];
    __shared__ float logits[NTYPES_];
    const int idx = topi[b * KMAX_ + k];
    ev[tid] = fusedf[((size_t)b * T_ + idx) * CC_ + tid];
    __syncthreads();

    // tokens = LN(GELU(ev @ tok_w + tok_b))
    float a = tokb[tid];
    for (int i = 0; i < CC_; ++i) a += ev[i] * tokw[(size_t)i * DM_ + tid];
    float g = gelu_exact(a);

    red[tid] = g; __syncthreads();
    for (int off = 128; off > 0; off >>= 1) { if (tid < off) red[tid] += red[tid + off]; __syncthreads(); }
    float mu = red[0] * (1.0f / 256.0f);
    __syncthreads();
    red[tid] = g * g; __syncthreads();
    for (int off = 128; off > 0; off >>= 1) { if (tid < off) red[tid] += red[tid + off]; __syncthreads(); }
    float var = red[0] * (1.0f / 256.0f) - mu * mu;
    const size_t ek = (size_t)b * KMAX_ + k;
    out[OFF_TOK + ek * DM_ + tid] = (g - mu) * rsqrtf(var + EPS_) * tlng[tid] + tlnb[tid];

    if (tid < NSCAL_) {
        float s = scb[tid];
        for (int i = 0; i < CC_; ++i) s += ev[i] * scw[i * NSCAL_ + tid];
        out[OFF_SC + ek * NSCAL_ + tid] = s;
    }
    if (tid < NTYPES_) {
        float s = tyb[tid];
        for (int i = 0; i < CC_; ++i) s += ev[i] * tyw[i * NTYPES_ + tid];
        logits[tid] = s;
    }
    __syncthreads();
    if (tid == 0) {
        int am = 0; float bb = logits[0];
        for (int j = 1; j < NTYPES_; ++j) if (logits[j] > bb) { bb = logits[j]; am = j; }
        out[OFF_TY + ek] = (float)am;
        float c = cfb[0];
        for (int i = 0; i < CC_; ++i) c += ev[i] * cfw[i];
        out[OFF_CF + ek] = sigmoid_f(c);
        out[OFF_EM + ek] = mask[(size_t)b * T_ + idx] ? 1.0f : 0.0f;
        out[OFF_TP + ek] = (float)idx;
        out[OFF_DE + ek] = 0.0f;
    }
}

// ---------------- host-side launch ----------------
extern "C" void kernel_launch(void* const* d_in, const int* in_sizes, int n_in,
                              void* d_out, int out_size, void* d_ws, size_t ws_size,
                              hipStream_t stream) {
    const float* feats = (const float*)d_in[0];
    const unsigned char* mask = (const unsigned char*)d_in[1];  // jax bool = 1 byte
    const float* conv_w[3] = {(const float*)d_in[2], (const float*)d_in[6], (const float*)d_in[10]};
    const float* conv_b[3] = {(const float*)d_in[3], (const float*)d_in[7], (const float*)d_in[11]};
    const float* gn_g[3]   = {(const float*)d_in[4], (const float*)d_in[8], (const float*)d_in[12]};
    const float* gn_b[3]   = {(const float*)d_in[5], (const float*)d_in[9], (const float*)d_in[13]};
    const float* fuse_w = (const float*)d_in[14]; const float* fuse_b = (const float*)d_in[15];
    const float* fln_g  = (const float*)d_in[16]; const float* fln_b  = (const float*)d_in[17];
    const float* gw1 = (const float*)d_in[18]; const float* gb1 = (const float*)d_in[19];
    const float* gw2 = (const float*)d_in[20]; const float* gb2 = (const float*)d_in[21];
    const float* tokw = (const float*)d_in[22]; const float* tokb = (const float*)d_in[23];
    const float* tlng = (const float*)d_in[24]; const float* tlnb = (const float*)d_in[25];
    const float* scw = (const float*)d_in[26]; const float* scb = (const float*)d_in[27];
    const float* tyw = (const float*)d_in[28]; const float* tyb = (const float*)d_in[29];
    const float* cfw = (const float*)d_in[30]; const float* cfb = (const float*)d_in[31];
    float* out = (float*)d_out;

    // workspace carve-out (256B aligned)
    char* ws = (char*)d_ws;
    size_t off = 0;
    auto carve = [&](size_t bytes) { size_t o = off; off = (off + bytes + 255) & ~(size_t)255; return o; };
    const size_t o_wconvT = carve((size_t)15 * CC_ * DIN_ * sizeof(bf16));
    const size_t o_wfuseT = carve((size_t)CC_ * 3 * CC_ * sizeof(bf16));
    const size_t o_wg1T   = carve((size_t)64 * CC_ * sizeof(bf16));
    const size_t o_Xbf    = carve((size_t)B_ * T_ * DIN_ * sizeof(bf16));
    const size_t o_Ybf    = carve((size_t)3 * B_ * T_ * CC_ * sizeof(bf16));
    const size_t o_stats  = carve((size_t)2 * 3 * B_ * 16 * sizeof(float));  // gsum|gsq
    const size_t o_fusedf = carve((size_t)B_ * T_ * CC_ * sizeof(float));
    const size_t o_fusedb = carve((size_t)B_ * T_ * CC_ * sizeof(bf16));
    const size_t o_scores = carve((size_t)B_ * T_ * sizeof(float));
    const size_t o_topi   = carve((size_t)B_ * KMAX_ * sizeof(int));
    const size_t o_tops   = carve((size_t)B_ * KMAX_ * sizeof(float));
    (void)ws_size; (void)in_sizes; (void)n_in; (void)out_size;

    bf16*  wconvT = (bf16*)(ws + o_wconvT);
    bf16*  wfuseT = (bf16*)(ws + o_wfuseT);
    bf16*  wg1T   = (bf16*)(ws + o_wg1T);
    bf16*  Xbf    = (bf16*)(ws + o_Xbf);
    bf16*  Ybf    = (bf16*)(ws + o_Ybf);
    float* gsum   = (float*)(ws + o_stats);
    float* gsq    = gsum + (size_t)3 * B_ * 16;
    float* fusedf = (float*)(ws + o_fusedf);
    bf16*  fusedbf= (bf16*)(ws + o_fusedb);
    float* scores = (float*)(ws + o_scores);
    int*   topi   = (int*)(ws + o_topi);
    float* tops   = (float*)(ws + o_tops);

    // zero GN stats (atomically accumulated)
    (void)hipMemsetAsync(gsum, 0, (size_t)2 * 3 * B_ * 16 * sizeof(float), stream);

    // pre-transforms
    {
        size_t nx = (size_t)B_ * T_ * DIN_;                     // multiple of 8*256
        k_x2bf<<<(unsigned)(nx / (8 * 256)), 256, 0, stream>>>(feats, Xbf);
        int nc = 15 * CC_ * DIN_;
        k_conv_wT<<<(nc + 255) / 256, 256, 0, stream>>>(conv_w[0], conv_w[1], conv_w[2], wconvT);
        int nf = CC_ * 3 * CC_;
        k_fuse_wT<<<(nf + 255) / 256, 256, 0, stream>>>(fuse_w, wfuseT);
        int ngw = 64 * CC_;
        k_g1_wT<<<(ngw + 255) / 256, 256, 0, stream>>>(gw1, wg1T);
    }

    // conv branches (implicit GEMM, WMMA, async double-buffered)
    k_conv_wmma<<<dim3(T_ / 64, B_, 3), 256, 0, stream>>>(
        Xbf, wconvT, conv_b[0], conv_b[1], conv_b[2], Ybf, gsum, gsq);

    // GN finalize + GELU
    {
        size_t total = (size_t)3 * B_ * T_ * CC_;
        k_gn_gelu<<<(unsigned)(total / (8 * 256)), 256, 0, stream>>>(
            Ybf, gsum, gsq, gn_g[0], gn_b[0], gn_g[1], gn_b[1], gn_g[2], gn_b[2]);
    }

    // fuse GEMM + GELU + LN
    k_fuse_wmma<<<dim3(T_ / 64, B_), 256, 0, stream>>>(
        Ybf, wfuseT, fuse_b, fln_g, fln_b, fusedf, fusedbf);

    // gate GEMM + sigmoid scores
    k_gate_wmma<<<dim3(T_ / 64, B_), 256, 0, stream>>>(
        fusedbf, wg1T, gb1, gw2, gb2, mask, scores);

    // top-k per batch
    k_topk<<<B_, 256, 0, stream>>>(scores, topi, tops);

    // event heads -> outputs
    k_head<<<dim3(KMAX_, B_), 256, 0, stream>>>(
        fusedf, topi, mask, tokw, tokb, tlng, tlnb, scw, scb, tyw, tyb, cfw, cfb, out);
}
